// GLAHead_62302795596530
// MI455X (gfx1250) — matrix operand order repaired and checked by reading, not compiled, not run on hardware
//
#include <hip/hip_runtime.h>

#define DM    1024
#define NQKVA 512
#define TSEQ  2048
#define NB    8

typedef __bf16 bf16_t;
typedef __bf16 v16bf __attribute__((ext_vector_type(16)));
typedef float  v8f   __attribute__((ext_vector_type(8)));
typedef float  f4    __attribute__((ext_vector_type(4)));
typedef unsigned int u4i __attribute__((ext_vector_type(4)));

// ---- TDM availability / arity probes (ROCm7.2 = 5 args, amdgpu-toolchain = 6) ----
#if defined(__has_builtin)
#if __has_builtin(__builtin_amdgcn_tensor_load_to_lds)
#define HAVE_TDM 1
#endif
#endif
#ifndef HAVE_TDM
#define HAVE_TDM 0
#endif
#if __has_include(<hip/amd_detail/amd_gfx1250_TDM.h>)
#define TDM_6ARG 1
#else
#define TDM_6ARG 0
#endif
// Compile-time proof that the device pass takes the TDM path (host pass exempt).
#if defined(__gfx1250__) && !HAVE_TDM
#error "gfx1250 device pass lacks __builtin_amdgcn_tensor_load_to_lds"
#endif

__device__ __forceinline__ unsigned short f2bf(float f) {
  return __builtin_bit_cast(unsigned short, (bf16_t)f);
}

// A-operand fragment (16x32 bf16) from row-major bf16 LDS.
// lane<16: row m=lane, K = {k0..k0+7, k0+16..k0+23}
// lane>=16: row m=lane-16, K = {k0+8..k0+15, k0+24..k0+31}
__device__ __forceinline__ v16bf fragA(const unsigned short* row, int k0, int hi) {
  const u4i* q = (const u4i*)(row + k0 + hi * 8);
  union { u4i u[2]; v16bf h; } f;
  f.u[0] = q[0];
  f.u[1] = q[2];
  return f.h;
}
// Same A pattern, f32 source, convert to bf16 at fragment build.
__device__ __forceinline__ v16bf fragA_f32(const float* row, int k0, int hi) {
  const f4* q = (const f4*)(row + k0 + hi * 8);
  f4 a = q[0], b = q[1], c = q[4], d = q[5];
  v16bf h;
  h[0]=(bf16_t)a.x; h[1]=(bf16_t)a.y; h[2]=(bf16_t)a.z;  h[3]=(bf16_t)a.w;
  h[4]=(bf16_t)b.x; h[5]=(bf16_t)b.y; h[6]=(bf16_t)b.z;  h[7]=(bf16_t)b.w;
  h[8]=(bf16_t)c.x; h[9]=(bf16_t)c.y; h[10]=(bf16_t)c.z; h[11]=(bf16_t)c.w;
  h[12]=(bf16_t)d.x; h[13]=(bf16_t)d.y; h[14]=(bf16_t)d.z; h[15]=(bf16_t)d.w;
  return h;
}
// B-operand fragment (32x16 bf16) from B^T stored row-major (row index = n).
// lane<16: col n=lane, K = {k0..k0+7, k0+16..k0+23} ; lane>=16: col n=lane-16, K = {k0+8..k0+15, k0+24..k0+31}
__device__ __forceinline__ v16bf fragB(const unsigned short* row, int k0, int hi) {
  const u4i* q = (const u4i*)(row + k0 + hi * 8);
  union { u4i u[2]; v16bf h; } f;
  f.u[0] = q[0];
  f.u[1] = q[2];
  return f.h;
}
// Same B pattern, f32 source.
__device__ __forceinline__ v16bf fragB_f32(const float* row, int k0, int hi) {
  const f4* q = (const f4*)(row + k0 + hi * 8);
  f4 a = q[0], b = q[1], c = q[4], d = q[5];
  v16bf h;
  h[0]=(bf16_t)a.x; h[1]=(bf16_t)a.y; h[2]=(bf16_t)a.z;  h[3]=(bf16_t)a.w;
  h[4]=(bf16_t)b.x; h[5]=(bf16_t)b.y; h[6]=(bf16_t)b.z;  h[7]=(bf16_t)b.w;
  h[8]=(bf16_t)c.x; h[9]=(bf16_t)c.y; h[10]=(bf16_t)c.z; h[11]=(bf16_t)c.w;
  h[12]=(bf16_t)d.x; h[13]=(bf16_t)d.y; h[14]=(bf16_t)d.z; h[15]=(bf16_t)d.w;
  return h;
}

__device__ __forceinline__ v8f wmma_bf16(v16bf a, v16bf b, v8f c) {
  return __builtin_amdgcn_wmma_f32_16x16x32_bf16(false, a, false, b, (short)0, c,
                                                 false, false);
}

// ---------------------------------------------------------------------------
// Tensor Data Mover: 2D tile load (global -> LDS) with LDS row padding.
// Descriptor packing per CDNA5 ISA 8.3/8.4 (D# group0 128b, group1 256b).
// ---------------------------------------------------------------------------
#if HAVE_TDM
typedef unsigned int tdm_u4 __attribute__((ext_vector_type(4)));
typedef int tdm_i8 __attribute__((ext_vector_type(8)));
typedef int tdm_i4 __attribute__((ext_vector_type(4)));

__device__ __forceinline__ void tdm_load_2d(unsigned lds_addr,
                                            unsigned long long gaddr,
                                            unsigned tile_d0, unsigned tile_d1,
                                            unsigned long long stride0_elems,
                                            unsigned dsz_code,   // 0=1B 1=2B 2=4B
                                            unsigned pad_int,    // 2^(n+1) dwords
                                            unsigned pad_amt) {  // (n+1) dwords
  tdm_u4 g0;
  g0[0] = 1u;                                              // count=1, user D#
  g0[1] = lds_addr;                                        // lds_addr (bytes)
  g0[2] = (unsigned)gaddr;                                 // global_addr[31:0]
  g0[3] = (unsigned)((gaddr >> 32) & 0x01FFFFFFull) | (2u << 30);  // [56:32]|type=2
  tdm_i8 g1;
  g1[0] = (int)((dsz_code << 16) | (1u << 20) | (pad_int << 22) | (pad_amt << 25));
  g1[1] = (int)(0u);                  // abar=0 | tensor_dim0[15:0]=0 (dim0 = 1<<30)
  g1[2] = (int)(0x4000u);             // tensor_dim0[31:16]=0x4000 | tensor_dim1[15:0]=0
  g1[3] = (int)(0x10u | (tile_d0 << 16));  // tensor_dim1[31:16]=0x10 (1<<20) | tile_dim0
  g1[4] = (int)(tile_d1);             // tile_dim1 | tile_dim2=0
  g1[5] = (int)(unsigned)stride0_elems;        // tensor_dim0_stride[31:0]
  g1[6] = (int)((unsigned)(stride0_elems >> 32) & 0xFFFFu);  // stride[47:32] | d1s=0
  g1[7] = 0;
  tdm_i4 z4 = {0, 0, 0, 0};
#if TDM_6ARG
  tdm_i8 z8 = {0, 0, 0, 0, 0, 0, 0, 0};
  __builtin_amdgcn_tensor_load_to_lds(g0, g1, z4, z4, z8, 0);
#else
  __builtin_amdgcn_tensor_load_to_lds(g0, g1, z4, z4, 0);
#endif
}
#endif  // HAVE_TDM

// ---------------------------------------------------------------------------
// Kernel 0: pack W_q|W_k|W_v|W_alpha (each 1024x128 f32) into bf16 W^T [512][1024]
// ---------------------------------------------------------------------------
__global__ __launch_bounds__(256) void prep_weights(
    const float* __restrict__ Wq, const float* __restrict__ Wk,
    const float* __restrict__ Wv, const float* __restrict__ Wa,
    unsigned short* __restrict__ wt) {
  int idx = blockIdx.x * 256 + threadIdx.x;   // 0 .. 512*1024-1
  int n = idx >> 10;
  int k = idx & 1023;
  const float* W = (n < 128) ? Wq : (n < 256) ? Wk : (n < 384) ? Wv : Wa;
  wt[(size_t)n * DM + k] = f2bf(W[(size_t)k * 128 + (n & 127)]);
}

// ---------------------------------------------------------------------------
// Kernel 1: QKVA = sigmoid-fused GEMM: [16384x1024] x [1024x512] -> f32
// grid (128,2); block 256 (8 waves). Double-buffered LDS fed by TDM:
//   wave0: x tile (128 x 32 f32, row pad 32dw+4dw), wave1: W^T tile (256 x 32 bf16).
// ---------------------------------------------------------------------------
#define XA_STR 36   // f32 row stride (32 + 4 pad dwords)
#define WS_STR 40   // bf16 row stride (32 + 8 pad halves = 16 + 4 pad dwords)

__global__ __launch_bounds__(256) void proj_kernel(
    const float* __restrict__ x, const unsigned short* __restrict__ wt,
    const float* __restrict__ b_alpha, float* __restrict__ qkva) {
  __shared__ __align__(16) float          XS[2][128 * XA_STR];
  __shared__ __align__(16) unsigned short WS[2][256 * WS_STR];
  const int tid = threadIdx.x, lane = tid & 31, wave = tid >> 5;
  const int lr = lane & 15, hi = lane >> 4;
  const int bm = blockIdx.x;    // 128 row blocks of 128
  const int bn = blockIdx.y;    // 2 col blocks of 256

  v8f zero = {};
  v8f acc[16];
#pragma unroll
  for (int nt = 0; nt < 16; ++nt) acc[nt] = zero;

  const float* xblk = x + (size_t)bm * 128 * DM;
  const unsigned short* wblk = wt + (size_t)bn * 256 * DM;

#if HAVE_TDM
  if (wave == 0)
    tdm_load_2d((unsigned)(size_t)&XS[0][0], (unsigned long long)(size_t)xblk,
                32, 128, DM, /*f32*/ 2, /*pad@32dw*/ 4, /*+4dw*/ 3);
  if (wave == 1)
    tdm_load_2d((unsigned)(size_t)&WS[0][0], (unsigned long long)(size_t)wblk,
                32, 256, DM, /*bf16*/ 1, /*pad@16dw*/ 3, /*+4dw*/ 3);
#endif

  int cur = 0;
  for (int k0 = 0; k0 < DM; k0 += 32) {
    const int nxt = cur ^ 1;
#if HAVE_TDM
    if (k0 + 32 < DM) {
      if (wave == 0)
        tdm_load_2d((unsigned)(size_t)&XS[nxt][0],
                    (unsigned long long)(size_t)(xblk + k0 + 32),
                    32, 128, DM, 2, 4, 3);
      if (wave == 1)
        tdm_load_2d((unsigned)(size_t)&WS[nxt][0],
                    (unsigned long long)(size_t)(wblk + k0 + 32),
                    32, 256, DM, 1, 3, 3);
      if (wave < 2) __builtin_amdgcn_s_wait_tensorcnt(1);  // current buffer done
    } else {
      if (wave < 2) __builtin_amdgcn_s_wait_tensorcnt(0);
    }
    __syncthreads();
#else
    __syncthreads();
    {  // fallback staging through VGPRs (buffer 0 only)
      const int m = tid >> 1, part = tid & 1;
      const f4* src = (const f4*)(xblk + (size_t)m * DM + k0 + part * 16);
      f4* dst = (f4*)&XS[0][m * XA_STR + part * 16];
#pragma unroll
      for (int i = 0; i < 4; ++i) dst[i] = src[i];
      const u4i* ws = (const u4i*)(wblk + (size_t)tid * DM + k0);
      u4i* wd = (u4i*)&WS[0][tid * WS_STR];
#pragma unroll
      for (int i = 0; i < 4; ++i) wd[i] = ws[i];
    }
    __syncthreads();
#endif
    // A fragment once per k-step; B fragments software-pipelined one tile ahead
    // so the ds_load_b128 pair for nt+1 issues under wmma(nt).
    v16bf a = fragA_f32(&XS[cur][(wave * 16 + lr) * XA_STR], 0, hi);
    v16bf b = fragB(&WS[cur][(0 * 16 + lr) * WS_STR], 0, hi);
#pragma unroll
    for (int nt = 0; nt < 16; ++nt) {
      v16bf bnext;
      if (nt < 15) bnext = fragB(&WS[cur][((nt + 1) * 16 + lr) * WS_STR], 0, hi);
      acc[nt] = wmma_bf16(a, b, acc[nt]);
      b = bnext;
    }
    __syncthreads();  // compute done before this buffer is re-filled
#if HAVE_TDM
    cur = nxt;
#endif
  }

  for (int nt = 0; nt < 16; ++nt) {
    const int col = bn * 256 + nt * 16 + lr;
    const bool is_alpha = (col >= 384);
    const float bias = is_alpha ? b_alpha[col - 384] : 0.f;
#pragma unroll
    for (int r = 0; r < 8; ++r) {
      const int m = r + 8 * hi;
      float v = acc[nt][r];
      if (is_alpha) v = 1.f / (1.f + __expf(-(v + bias)));
      qkva[(size_t)(bm * 128 + wave * 16 + m) * NQKVA + col] = v;
    }
  }
}

// ---------------------------------------------------------------------------
// Kernel 2: chunked gated scan. grid 32 = (batch, v-slice of 32); block 256.
// Per chunk (L=64):  P = cumprod(alpha);  q~=q*P;  k~=k/P;
//   O = q~ @ S + tril(q~ @ k~^T) @ V ;  S = diag(P_L) * (S + k~^T @ V)
// ---------------------------------------------------------------------------
__global__ __launch_bounds__(256) void scan_kernel(
    const float* __restrict__ qkva, float* __restrict__ out) {
  __shared__ __align__(16) float ST[32 * 132];            // S^T [v][k] f32, persists
  __shared__ __align__(16) float P[64 * 132];             // gate prefix products
  __shared__ float PL[128];                               // P at t = L-1
  __shared__ __align__(16) unsigned short Qt[64 * 136];   // q~  [t][k] bf16
  __shared__ __align__(16) unsigned short Kt[64 * 136];   // k~  [t][k] bf16
  __shared__ __align__(16) unsigned short KHT[128 * 72];  // k~^T [k][t] bf16
  __shared__ __align__(16) unsigned short VT[32 * 72];    // V^T slice [v][t] bf16
  __shared__ __align__(16) unsigned short ATT[64 * 72];   // masked attn [t][i] bf16

  const int tid = threadIdx.x, lane = tid & 31, wave = tid >> 5;
  const int lr = lane & 15, hi = lane >> 4;
  const int b = blockIdx.x >> 2;
  const int vbase = (blockIdx.x & 3) * 32;
  const int tm1 = wave & 3, tn1 = wave >> 2;        // m1/m3 tile
  const int tm2 = wave >> 1, tnb = (wave & 1) * 2;  // m2 tiles
  v8f zero = {};

  for (int e = tid; e < 32 * 132; e += 256) ST[e] = 0.f;
  __syncthreads();

  for (int c = 0; c < 32; ++c) {
    const size_t row0 = (size_t)b * TSEQ + (size_t)c * 64;
    if (c + 1 < 32) {  // speculative prefetch of next chunk (global_prefetch_b8)
      __builtin_prefetch(qkva + (row0 + 64 + (tid >> 3)) * NQKVA + (tid & 7) * 64, 0, 1);
    }
    // 1) stage gates
#pragma unroll 4
    for (int i = 0; i < 32; ++i) {
      int e = tid + 256 * i, t = e >> 7, k = e & 127;
      P[t * 132 + k] = qkva[(row0 + t) * NQKVA + 384 + k];
    }
    __syncthreads();
    // 2) in-chunk prefix product along t (one thread per k dim)
    if (tid < 128) {
      float p = 1.f;
      for (int t = 0; t < 64; ++t) { p *= P[t * 132 + tid]; P[t * 132 + tid] = p; }
      PL[tid] = p;
    }
    __syncthreads();
    // 3) build bf16 chunk operands
#pragma unroll 4
    for (int i = 0; i < 32; ++i) {
      int e = tid + 256 * i, t = e >> 7, k = e & 127;
      float p = P[t * 132 + k];
      float q = qkva[(row0 + t) * NQKVA + k];
      float kk = qkva[(row0 + t) * NQKVA + 128 + k] / p;
      Qt[t * 136 + k] = f2bf(q * p);
      unsigned short kb = f2bf(kk);
      Kt[t * 136 + k] = kb;
      KHT[k * 72 + t] = kb;
    }
#pragma unroll 4
    for (int i = 0; i < 8; ++i) {
      int e = tid + 256 * i, t = e >> 5, v = e & 31;
      VT[v * 72 + t] = f2bf(qkva[(row0 + t) * NQKVA + 256 + vbase + v]);
    }
    __syncthreads();

    // m1: O_inter = q~ @ S   (64x128 @ 128x32), one 16x16 tile per wave
    v8f acc_o = zero;
#pragma unroll
    for (int k0 = 0; k0 < 128; k0 += 32) {
      v16bf a = fragA(Qt + (tm1 * 16 + lr) * 136, k0, hi);
      v16bf bb = fragB_f32(ST + (tn1 * 16 + lr) * 132, k0, hi);
      acc_o = wmma_bf16(a, bb, acc_o);
    }
    // m2: Att = q~ @ k~^T   (64x64), two tiles per wave
    v8f att0 = zero, att1 = zero;
#pragma unroll
    for (int k0 = 0; k0 < 128; k0 += 32) {
      v16bf a = fragA(Qt + (tm2 * 16 + lr) * 136, k0, hi);
      att0 = wmma_bf16(a, fragB(Kt + ((tnb + 0) * 16 + lr) * 136, k0, hi), att0);
      att1 = wmma_bf16(a, fragB(Kt + ((tnb + 1) * 16 + lr) * 136, k0, hi), att1);
    }
    // causal mask + store to LDS
#pragma unroll
    for (int r = 0; r < 8; ++r) {
      int t = tm2 * 16 + r + 8 * hi;
      int i0 = tnb * 16 + lr, i1 = i0 + 16;
      ATT[t * 72 + i0] = f2bf(i0 <= t ? att0[r] : 0.f);
      ATT[t * 72 + i1] = f2bf(i1 <= t ? att1[r] : 0.f);
    }
    __syncthreads();
    // m3: O += tril(Att) @ V   (64x64 @ 64x32)
#pragma unroll
    for (int k0 = 0; k0 < 64; k0 += 32) {
      v16bf a = fragA(ATT + (tm1 * 16 + lr) * 72, k0, hi);
      v16bf bb = fragB(VT + (tn1 * 16 + lr) * 72, k0, hi);
      acc_o = wmma_bf16(a, bb, acc_o);
    }
    // m4: dS = k~^T @ V   (128x64 @ 64x32), two tiles per wave (tm=wave)
    v8f ds0 = zero, ds1 = zero;
#pragma unroll
    for (int k0 = 0; k0 < 64; k0 += 32) {
      v16bf a = fragA(KHT + (wave * 16 + lr) * 72, k0, hi);
      ds0 = wmma_bf16(a, fragB(VT + (0 * 16 + lr) * 72, k0, hi), ds0);
      ds1 = wmma_bf16(a, fragB(VT + (1 * 16 + lr) * 72, k0, hi), ds1);
    }
    __syncthreads();  // all m1 reads of ST complete before update
    // state update: S^T[v][k] = PL[k] * (S^T[v][k] + dS[k][v])
#pragma unroll
    for (int r = 0; r < 8; ++r) {
      int k = wave * 16 + r + 8 * hi;
      float pl = PL[k];
      ST[lr * 132 + k]        = pl * (ST[lr * 132 + k] + ds0[r]);
      ST[(16 + lr) * 132 + k] = pl * (ST[(16 + lr) * 132 + k] + ds1[r]);
    }
    // output store
#pragma unroll
    for (int r = 0; r < 8; ++r) {
      int t = tm1 * 16 + r + 8 * hi;
      int v = tn1 * 16 + lr;
      out[((size_t)b * TSEQ + c * 64 + t) * 128 + vbase + v] = acc_o[r];
    }
    __syncthreads();
  }
}

// ---------------------------------------------------------------------------
extern "C" void kernel_launch(void* const* d_in, const int* in_sizes, int n_in,
                              void* d_out, int out_size, void* d_ws, size_t ws_size,
                              hipStream_t stream) {
  const float* x  = (const float*)d_in[0];
  const float* Wq = (const float*)d_in[1];
  const float* Wk = (const float*)d_in[2];
  const float* Wv = (const float*)d_in[3];
  const float* Wa = (const float*)d_in[4];
  const float* ba = (const float*)d_in[5];
  float* out = (float*)d_out;

  float* qkva = (float*)d_ws;                                   // 16384*512*4 = 32 MB
  unsigned short* wt =
      (unsigned short*)((char*)d_ws + (size_t)16384 * NQKVA * sizeof(float));  // +1 MB

  prep_weights<<<(512 * 1024) / 256, 256, 0, stream>>>(Wq, Wk, Wv, Wa, wt);
  proj_kernel<<<dim3(128, 2), 256, 0, stream>>>(x, wt, ba, qkva);
  scan_kernel<<<32, 256, 0, stream>>>(qkva, out);
}
